// RelationEmbeddingNet_19816979103961
// MI455X (gfx1250) — matrix operand order repaired and checked by reading, compiled, not run on hardware
//
#include <hip/hip_runtime.h>
#include <math.h>

typedef _Float16 v16h  __attribute__((ext_vector_type(16)));
typedef float    v8f   __attribute__((ext_vector_type(8)));
typedef unsigned u32x8 __attribute__((ext_vector_type(8)));

#define B_   64
#define T_   256
#define NOBJ 8192
#define DE   32
#define DA   128

// hardware tanh (v_tanh_f32 on gfx1250); fallback is branch-free via v_exp_f32
__device__ __forceinline__ float fast_tanh(float x) {
#if __has_builtin(__builtin_amdgcn_tanhf)
    return __builtin_amdgcn_tanhf(x);
#else
    float e = __expf(2.0f * x);
    return 1.0f - 2.0f / (e + 1.0f);
#endif
}

__device__ __forceinline__ unsigned pack2h(float a, float b) {
    union { _Float16 h[2]; unsigned u; } uu;
    uu.h[0] = (_Float16)a; uu.h[1] = (_Float16)b;
    return uu.u;   // lowers to v_cvt_pk_f16_f32
}

// ---------------------------------------------------------------------------
// Kernel A: q = h_ped @ Wah + ba  [64,128]; ped_t/ego_t -> feats cols 0-31/192-223
// ---------------------------------------------------------------------------
__global__ __launch_bounds__(256) void prep_kernel(
    const float* __restrict__ x_ped, const float* __restrict__ x_ego,
    const float* __restrict__ h_ped,
    const float* __restrict__ Wp, const float* __restrict__ bp,
    const float* __restrict__ We, const float* __restrict__ be,
    const float* __restrict__ Wah, const float* __restrict__ ba,
    const int* __restrict__ t_ptr,
    float* __restrict__ q_out, float* __restrict__ out)
{
    const int t = *t_ptr;
    for (int idx = threadIdx.x; idx < B_ * DA; idx += 256) {
        int b = idx / DA, j = idx % DA;
        float v = ba[j];
        for (int k = 0; k < DE; ++k) v += h_ped[b * DE + k] * Wah[k * DA + j];
        q_out[idx] = v;
    }
    for (int idx = threadIdx.x; idx < B_ * DE; idx += 256) {
        int b = idx / DE, j = idx % DE;
        float vp = bp[j], ve = be[j];
        const float* xp = x_ped + (b * T_ + t) * 4;
        const float* xe = x_ego + (b * T_ + t) * 4;
        for (int k = 0; k < 4; ++k) {
            vp += xp[k] * Wp[k * DE + j];
            ve += xe[k] * We[k * DE + j];
        }
        out[64 + b * 224 + j]       = fmaxf(vp, 0.f);   // ped_t
        out[64 + b * 224 + 192 + j] = fmaxf(ve, 0.f);   // ego_t
    }
}

__global__ void zero_kernel(float* __restrict__ p, int n) {
    int i = blockIdx.x * 256 + threadIdx.x;
    if (i < n) p[i] = 0.f;
}

// ---------------------------------------------------------------------------
// Kernel B: fused per-type attention. 128 threads = 4 waves; each wave owns a
// 16-object M-tile. emb (d<=7 -> 32, relu) via VALU into LDS; emb@Wax via
// 8x v_wmma_f32_16x16x32_f16 (K=32 == D_EMB, fragments pre-packed in LDS);
// v_tanh_f32 + va-dot reduced via LDS; e = mask*exp(score); atomics
// accumulate denom[seg] and emb*e -> acc.
// ---------------------------------------------------------------------------
__global__ __launch_bounds__(128) void attend_kernel(
    const float* __restrict__ x, const int* __restrict__ cls, const int* __restrict__ seg,
    const float* __restrict__ W, const float* __restrict__ bvec, int d,
    const float* __restrict__ Wax, const float* __restrict__ va,
    const float* __restrict__ q, const int* __restrict__ t_ptr,
    float* __restrict__ e_out, float* __restrict__ denom, float* __restrict__ acc)
{
    // B fragments pre-packed per (tile, lane): 8 u32 = one v16h operand
    __shared__ __align__(32) unsigned sBfrag[8][32][8];   // 8 KB
    __shared__ float    sVa[DA];
    __shared__ float    sEmb [4][16][DE];                 // f32 emb (feat acc)  8 KB
    __shared__ _Float16 sEmbH[4][16][DE];                 // f16 emb (A frag)    4 KB
    __shared__ float    sRed[4][32][8];                   // score reduction     4 KB
    __shared__ float    sE[4][16];
    __shared__ int      sSeg[4][16];
    __shared__ int      sMask[4][16];

    const int t    = *t_ptr;
    const int tid  = threadIdx.x;
    const int w    = tid >> 5;
    const int lane = tid & 31;
    const int base = (blockIdx.x * 4 + w) * 16;

    // pack Wax into WMMA B-fragment layout:
    // word(nt,L,v) = {Wax[kk][col], Wax[kk+1][col]}, kk=(L>>4)*16+2v, col=nt*16+(L&15)
    for (int i = tid; i < 8 * 32 * 8; i += 128) {
        int nt = i >> 8, rem = i & 255;
        int L = rem >> 3, v = rem & 7;
        int col = nt * 16 + (L & 15);
        int kk = (L >> 4) * 16 + 2 * v;
        sBfrag[nt][L][v] = pack2h(Wax[kk * DA + col], Wax[(kk + 1) * DA + col]);
    }
    if (tid < DA) sVa[tid] = va[tid];

    if (lane < 16) {
        int obj = base + lane;
        sSeg[w][lane]  = seg[obj];
        sMask[w][lane] = (cls[obj * T_ + t] != -1) ? 1 : 0;
    }
    {   // emb: lane j handles column j for the wave's 16 objects
        const int j = lane;
        for (int m = 0; m < 16; ++m) {
            int obj = base + m;
            const float* xr = x + (obj * T_ + t) * d;
            float v = bvec[j];
            for (int k = 0; k < d; ++k) v += xr[k] * W[k * DE + j];
            v = fmaxf(v, 0.f);
            sEmb[w][m][j]  = v;
            sEmbH[w][m][j] = (_Float16)v;
        }
    }
    __syncthreads();

    // A fragment: 16x32 f16, K-pairs at even offsets -> 8 aligned u32 LDS loads
    const int Mrow = lane & 15, half = lane >> 4;
    const unsigned* arow = (const unsigned*)&sEmbH[w][Mrow][0];
    u32x8 aw;
#pragma unroll
    for (int v = 0; v < 8; ++v) {
        int pidx = v + 8 * half;   // kb>>1
        aw[v] = arow[pidx];
    }
    const v16h A = __builtin_bit_cast(v16h, aw);

    float partial[8];
#pragma unroll
    for (int v = 0; v < 8; ++v) partial[v] = 0.f;

#pragma unroll
    for (int nt = 0; nt < 8; ++nt) {
        const int col = nt * 16 + Mrow;
        u32x8 bw;
        const unsigned* bp8 = &sBfrag[nt][lane][0];
#pragma unroll
        for (int v = 0; v < 8; ++v) bw[v] = bp8[v];
        const v16h Bf = __builtin_bit_cast(v16h, bw);

        v8f c = {};
        c = __builtin_amdgcn_wmma_f32_16x16x32_f16(false, A, false, Bf,
                                                   (short)0, c, false, false);
        const float vav = sVa[col];
#pragma unroll
        for (int v = 0; v < 8; ++v) {
            int M = v + 8 * half;                      // C/D: VGPR v <-> row M
            float s = c[v] + q[sSeg[w][M] * DA + col]; // + (h_ped@Wah+ba)[seg]
            partial[v] += fast_tanh(s) * vav;
        }
    }
#pragma unroll
    for (int v = 0; v < 8; ++v) sRed[w][lane][v] = partial[v];
    __syncthreads();

    if (lane < 16) {
        const int M = lane;
        const int gbase = (M >= 8) ? 16 : 0;
        const int v = M & 7;
        float s = 0.f;
        for (int l = 0; l < 16; ++l) s += sRed[w][gbase + l][v];
        // |score| <= sum|va| ~ 9 -> exp never overflows; max-subtraction not needed
        float e = sMask[w][M] ? __expf(s) : 0.f;
        sE[w][M] = e;
        e_out[base + M] = e;
        if (e != 0.f) atomicAdd(&denom[sSeg[w][M]], e);
    }
    __syncthreads();

    {   // feat accumulation: acc[seg] += emb * e (normalized later by denom)
        const int j = lane;
        for (int m = 0; m < 16; ++m) {
            float ev = sE[w][m];
            if (ev != 0.f) atomicAdd(&acc[sSeg[w][m] * DE + j], sEmb[w][m][j] * ev);
        }
    }
}

// p[n] = e[n] / max(denom[seg[n]], 1e-30)
__global__ void prob_kernel(const float* __restrict__ e, const float* __restrict__ denom,
                            const int* __restrict__ seg, float* __restrict__ p)
{
    int n = blockIdx.x * 256 + threadIdx.x;
    if (n < NOBJ) p[n] = e[n] / fmaxf(denom[seg[n]], 1e-30f);
}

// feats[b, 32+32*typ+j] = acc[typ][b][j] / max(denom[typ][b], 1e-30)
__global__ void feat_kernel(const float* __restrict__ acc, const float* __restrict__ denom,
                            float* __restrict__ out)
{
    int idx = blockIdx.x * 256 + threadIdx.x;
    if (idx < 5 * B_ * DE) {
        int typ = idx / (B_ * DE);
        int r = idx % (B_ * DE);
        int b = r / DE, j = r % DE;
        out[64 + b * 224 + 32 + 32 * typ + j] = acc[idx] / fmaxf(denom[typ * B_ + b], 1e-30f);
    }
}

// classifier head: relu(feats@W1+b1) -> relu(@W2+b2) -> @W3+b3 (64x1)
__global__ __launch_bounds__(256) void head_kernel(
    const float* __restrict__ W1, const float* __restrict__ b1,
    const float* __restrict__ W2, const float* __restrict__ b2,
    const float* __restrict__ W3, const float* __restrict__ b3,
    float* __restrict__ out)
{
    __shared__ float sH1[B_][DE];
    __shared__ float sH2[B_][DE];
    const float* feats = out + 64;
    for (int idx = threadIdx.x; idx < B_ * DE; idx += 256) {
        int b = idx / DE, j = idx % DE;
        float v = b1[j];
        for (int k = 0; k < 224; ++k) v += feats[b * 224 + k] * W1[k * DE + j];
        sH1[b][j] = fmaxf(v, 0.f);
    }
    __syncthreads();
    for (int idx = threadIdx.x; idx < B_ * DE; idx += 256) {
        int b = idx / DE, j = idx % DE;
        float v = b2[j];
        for (int k = 0; k < DE; ++k) v += sH1[b][k] * W2[k * DE + j];
        sH2[b][j] = fmaxf(v, 0.f);
    }
    __syncthreads();
    if (threadIdx.x < B_) {
        int b = threadIdx.x;
        float v = b3[0];
        for (int k = 0; k < DE; ++k) v += sH2[b][k] * W3[k];
        out[b] = v;
    }
}

// ---------------------------------------------------------------------------
extern "C" void kernel_launch(void* const* d_in, const int* in_sizes, int n_in,
                              void* d_out, int out_size, void* d_ws, size_t ws_size,
                              hipStream_t stream)
{
    const float* x_ped = (const float*)d_in[0];
    const float* x_ego = (const float*)d_in[1];
    const float* h_ped = (const float*)d_in[2];
    const float* x_t[5]; const int* cls_t[5]; const int* seg_t[5];
    for (int i = 0; i < 5; ++i) {
        x_t[i]   = (const float*)d_in[3 + 3 * i];
        cls_t[i] = (const int*)  d_in[4 + 3 * i];
        seg_t[i] = (const int*)  d_in[5 + 3 * i];
    }
    const float* Wtyp[5] = {(const float*)d_in[20], (const float*)d_in[22],
                            (const float*)d_in[24], (const float*)d_in[26],
                            (const float*)d_in[28]};
    const float* btyp[5] = {(const float*)d_in[21], (const float*)d_in[23],
                            (const float*)d_in[25], (const float*)d_in[27],
                            (const float*)d_in[29]};
    const int dtyp[5] = {4, 6, 5, 7, 7};   // neighbor, light, sign, crosswalk, station
    const float* Wp  = (const float*)d_in[18]; const float* bp = (const float*)d_in[19];
    const float* We  = (const float*)d_in[30]; const float* be = (const float*)d_in[31];
    const float* Wax = (const float*)d_in[32];
    const float* Wah = (const float*)d_in[33]; const float* ba = (const float*)d_in[34];
    const float* va  = (const float*)d_in[35];
    const float* W1  = (const float*)d_in[36]; const float* b1 = (const float*)d_in[37];
    const float* W2  = (const float*)d_in[38]; const float* b2 = (const float*)d_in[39];
    const float* W3  = (const float*)d_in[40]; const float* b3 = (const float*)d_in[41];
    const int*   t_ptr = (const int*)d_in[42];

    float* out = (float*)d_out;
    float* ws  = (float*)d_ws;
    float* q     = ws;            // 64*128      = 8192 floats
    float* e_buf = ws + 8192;     // 5*8192      = 40960
    float* denom = ws + 49152;    // 5*64        = 320
    float* acc   = ws + 49472;    // 5*64*32     = 10240  (contiguous with denom)

    prep_kernel<<<1, 256, 0, stream>>>(x_ped, x_ego, h_ped, Wp, bp, We, be,
                                       Wah, ba, t_ptr, q, out);
    const int nz = 320 + 10240;   // denom + acc (contiguous)
    zero_kernel<<<(nz + 255) / 256, 256, 0, stream>>>(denom, nz);
    for (int typ = 0; typ < 5; ++typ) {
        attend_kernel<<<NOBJ / 64, 128, 0, stream>>>(
            x_t[typ], cls_t[typ], seg_t[typ], Wtyp[typ], btyp[typ], dtyp[typ],
            Wax, va, q, t_ptr,
            e_buf + typ * NOBJ, denom + typ * 64, acc + typ * 64 * DE);
    }
    for (int typ = 0; typ < 5; ++typ) {
        prob_kernel<<<NOBJ / 256, 256, 0, stream>>>(
            e_buf + typ * NOBJ, denom + typ * 64, seg_t[typ],
            out + 14400 + typ * NOBJ);
    }
    feat_kernel<<<(5 * B_ * DE + 255) / 256, 256, 0, stream>>>(acc, denom, out);
    head_kernel<<<1, 256, 0, stream>>>(W1, b1, W2, b2, W3, b3, out);
}
